// EncoderLayer_70755291235067
// MI455X (gfx1250) — matrix operand order repaired and checked by reading, not compiled, not run on hardware
//
#include <hip/hip_runtime.h>
#include <hip/hip_bf16.h>
#include <math.h>

// ---------------------------------------------------------------------------
// Problem constants (match reference)
// ---------------------------------------------------------------------------
#define B_   8
#define N_   1024
#define DI_  64
#define D_   128
#define H_   8
#define DK_  32
#define L_   6
#define M_   (B_ * N_)     // 8192 rows
#define HDK_ (H_ * DK_)    // 256
#define SCALE_ 0.17677669529663687f   // 1/sqrt(32)

typedef __bf16 bf16;
typedef bf16 v8bf  __attribute__((ext_vector_type(8)));
typedef bf16 v16bf __attribute__((ext_vector_type(16)));
typedef float v8f  __attribute__((ext_vector_type(8)));

// Async global->LDS DMA path (gfx1250), guarded so compile never breaks.
#if defined(__has_builtin)
#if __has_builtin(__builtin_amdgcn_global_load_async_to_lds_b128) && \
    __has_builtin(__builtin_amdgcn_s_wait_asynccnt)
#define HAVE_ASYNC_LDS 1
#endif
#endif

#ifdef HAVE_ASYNC_LDS
// Builtin expects pointers to 16-byte int vectors (per clang diagnostic:
// '__attribute__((__vector_size__(4 * sizeof(int)))) int __device__ *').
typedef int v4i_ __attribute__((vector_size(4 * sizeof(int))));
typedef __attribute__((address_space(1))) v4i_* gp128;
typedef __attribute__((address_space(3))) v4i_* lp128;
#endif

// ---------------------------------------------------------------------------
// WMMA helpers (CDNA5 wave32, D = A(16x32) * B(32x16) + C, f32 accumulate)
// A layout: lane = row(0..15) in both halves; per-lane 16 contiguous K at
//           half*16 (row-major A).
// B layout: lane = col(0..15) in both halves; per-lane 16 contiguous K at
//           half*16 when B is stored column-major ([n][k]).
// C layout: VGPR g -> row = g + half*8, col = lane&15.
// ---------------------------------------------------------------------------
__device__ __forceinline__ v8f wmma_bf(v16bf a, v16bf b, v8f c) {
  return __builtin_amdgcn_wmma_f32_16x16x32_bf16(false, a, false, b,
                                                 (short)0, c, false, false);
}

__device__ __forceinline__ v16bf load_a(const bf16* src, int ld, int lane) {
  int row = lane & 15, hlf = lane >> 4;
  return *(const v16bf*)(src + (size_t)row * ld + hlf * 16);
}

__device__ __forceinline__ v16bf load_b(const bf16* srcT, int ld, int lane) {
  int n = lane & 15, hlf = lane >> 4;
  return *(const v16bf*)(srcT + (size_t)n * ld + hlf * 16);
}

// Butterfly reductions over the 16-lane half-group (xor bits 0..3 only).
__device__ __forceinline__ float bfly_add(float v) {
#pragma unroll
  for (int off = 1; off <= 8; off <<= 1) v += __shfl_xor(v, off, 32);
  return v;
}
__device__ __forceinline__ float bfly_max(float v) {
#pragma unroll
  for (int off = 1; off <= 8; off <<= 1) v = fmaxf(v, __shfl_xor(v, off, 32));
  return v;
}

// ---------------------------------------------------------------------------
// Prep kernels
// ---------------------------------------------------------------------------
__global__ __launch_bounds__(256) void k_cvt_bf16(const float* __restrict__ s,
                                                  bf16* __restrict__ d, int n) {
  int i = blockIdx.x * 256 + threadIdx.x;
  if (i < n) d[i] = (bf16)s[i];
}

// Wt[n*K + k] = W[k*Ncol + n]  (f32 row-major [K][Ncol] -> bf16 col-major)
__global__ __launch_bounds__(256) void k_trans_bf16(const float* __restrict__ W,
                                                    bf16* __restrict__ Wt,
                                                    int K, int Ncol) {
  int i = blockIdx.x * 256 + threadIdx.x;
  if (i < K * Ncol) {
    int n = i / K, k = i - n * K;
    Wt[i] = (bf16)W[(size_t)k * Ncol + n];
  }
}

// adj int32 -> bitmask, word i covers adj[i*32 .. i*32+31]
__global__ __launch_bounds__(256) void k_pack_mask(const int* __restrict__ adj,
                                                   unsigned* __restrict__ Mg,
                                                   int total) {
  int i = blockIdx.x * 256 + threadIdx.x;
  if (i < total) {
    const int* p = adj + (size_t)i * 32;
    unsigned w = 0;
#pragma unroll
    for (int j = 0; j < 32; ++j) w |= (p[j] != 0 ? 1u : 0u) << j;
    Mg[i] = w;
  }
}

// ---------------------------------------------------------------------------
// Generic GEMM: C = A[M,Ka](bf16) * Bt[Ncol,Ka](bf16) + bias, optional ReLU.
// One wave per 16x64 output block: each A operand feeds 4 WMMAs.
// ---------------------------------------------------------------------------
__global__ __launch_bounds__(256) void k_gemm_bias(
    const bf16* __restrict__ A, const bf16* __restrict__ Bt,
    const float* __restrict__ bias, float* __restrict__ Cf,
    bf16* __restrict__ Cb, int M, int Ka, int Ncol, int relu) {
  int wid = blockIdx.x * 8 + (threadIdx.x >> 5);
  int lane = threadIdx.x & 31;
  int NG = Ncol >> 6;                    // groups of 64 columns
  int mt = wid / NG, cg = wid - mt * NG;
  if (mt * 16 >= M) return;
  const v8f zf = {0, 0, 0, 0, 0, 0, 0, 0};
  v8f c[4];
#pragma unroll
  for (int t = 0; t < 4; ++t) c[t] = zf;
  const bf16* Bbase = Bt + (size_t)cg * 64 * Ka;
  for (int k0 = 0; k0 < Ka; k0 += 32) {
    v16bf a = load_a(A + (size_t)mt * 16 * Ka + k0, Ka, lane);
#pragma unroll
    for (int t = 0; t < 4; ++t)
      c[t] = wmma_bf(a, load_b(Bbase + (size_t)t * 16 * Ka + k0, Ka, lane),
                     c[t]);
  }
  int hlf = lane >> 4, n = lane & 15;
#pragma unroll
  for (int t = 0; t < 4; ++t) {
    int col = cg * 64 + t * 16 + n;
    float bv = bias ? bias[col] : 0.0f;
#pragma unroll
    for (int g = 0; g < 8; ++g) {
      int row = mt * 16 + g + hlf * 8;
      float v = c[t][g] + bv;
      if (relu) v = fmaxf(v, 0.0f);
      if (Cf) Cf[(size_t)row * Ncol + col] = v;
      if (Cb) Cb[(size_t)row * Ncol + col] = (bf16)v;
    }
  }
}

// ---------------------------------------------------------------------------
// Fused QKV projection: q/k/v share the A operand (one A load -> 3 WMMAs).
// Scatter into attention-friendly layouts:
//   Q,K: [b][h][n][dk] bf16 (Q pre-scaled by 1/sqrt(dk))
//   V  : transposed [b][h][dk][n] bf16 (B-operand for O = P*V)
// ---------------------------------------------------------------------------
__global__ __launch_bounds__(256) void k_qkv(
    const bf16* __restrict__ Hb, const bf16* __restrict__ WqT,
    const bf16* __restrict__ WkT, const bf16* __restrict__ WvT,
    bf16* __restrict__ Qb, bf16* __restrict__ Kb, bf16* __restrict__ Vt) {
  int wid = blockIdx.x * 8 + (threadIdx.x >> 5);
  int lane = threadIdx.x & 31;
  const int NT = HDK_ / 16;  // 16
  int mt = wid / NT, ct = wid - mt * NT;
  const v8f zf = {0, 0, 0, 0, 0, 0, 0, 0};
  v8f cq = zf, ck = zf, cv = zf;
  size_t wb = (size_t)ct * 16 * D_;
  for (int k0 = 0; k0 < D_; k0 += 32) {
    v16bf a = load_a(Hb + (size_t)mt * 16 * D_ + k0, D_, lane);
    cq = wmma_bf(a, load_b(WqT + wb + k0, D_, lane), cq);
    ck = wmma_bf(a, load_b(WkT + wb + k0, D_, lane), ck);
    cv = wmma_bf(a, load_b(WvT + wb + k0, D_, lane), cv);
  }
  int hlf = lane >> 4, n = lane & 15;
  int col = ct * 16 + n, hd = col >> 5, dk = col & 31;
#pragma unroll
  for (int g = 0; g < 8; ++g) {
    int row = mt * 16 + g + hlf * 8;
    int bb = row >> 10, nn = row & (N_ - 1);
    Qb[(((size_t)bb * H_ + hd) * N_ + nn) * DK_ + dk] = (bf16)(cq[g] * SCALE_);
    Kb[(((size_t)bb * H_ + hd) * N_ + nn) * DK_ + dk] = (bf16)ck[g];
    Vt[(((size_t)bb * H_ + hd) * DK_ + dk) * N_ + nn] = (bf16)cv[g];
  }
}

// ---------------------------------------------------------------------------
// Flash attention. One WG per (b,h, 128 query rows); 8 waves x 16 rows.
// LDS: K[1024][32] bf16 (64KB) + Vt[32][1024] bf16 (64KB)
//      + P scratch [8][16][32] bf16 (8KB) + mask bits [128][32] u32 (16KB)
// = 152 KB of the 320 KB/WGP.  K/V/mask staged by async DMA when available.
// ---------------------------------------------------------------------------
#define ATT_LDS_BYTES ((1024 * 32 + 32 * 1024 + 8 * 16 * 32) * 2 + 128 * 32 * 4)

__global__ __launch_bounds__(256) void k_attn(
    const bf16* __restrict__ Qg, const bf16* __restrict__ Kg,
    const bf16* __restrict__ Vg, const unsigned* __restrict__ Mg,
    float* __restrict__ Of, bf16* __restrict__ Ob) {
  extern __shared__ char smem[];
  bf16* Ksh = (bf16*)smem;               // [1024][32]  key-major (= col-major K^T)
  bf16* Vsh = Ksh + 1024 * 32;           // [32][1024]  dk-major  (= col-major V)
  bf16* Psh = Vsh + 32 * 1024;           // [8 waves][16][32]
  unsigned* Msh = (unsigned*)(Psh + 8 * 16 * 32);  // [128][32]

  int wg = blockIdx.x;
  int qblk = wg & 7;        // N_/128 = 8
  int bh = wg >> 3;         // b*H + h
  int b = bh >> 3;          // H_ = 8
  int hd = bh & 7;
  int qbase = qblk * 128;
  int tid = threadIdx.x;

  const char* kc = (const char*)(Kg + (size_t)bh * N_ * DK_);
  const char* vc = (const char*)(Vg + (size_t)bh * DK_ * N_);
  const char* mc = (const char*)(Mg + ((size_t)b * N_ + qbase) * (N_ / 32));

#ifdef HAVE_ASYNC_LDS
  for (int i = tid; i < (N_ * DK_ * 2) / 16; i += 256)        // 64KB of K
    __builtin_amdgcn_global_load_async_to_lds_b128(
        (gp128)(v4i_*)(kc + (size_t)i * 16),
        (lp128)(v4i_*)((char*)Ksh + (size_t)i * 16), 0, 0);
  for (int i = tid; i < (DK_ * N_ * 2) / 16; i += 256)        // 64KB of V^T
    __builtin_amdgcn_global_load_async_to_lds_b128(
        (gp128)(v4i_*)(vc + (size_t)i * 16),
        (lp128)(v4i_*)((char*)Vsh + (size_t)i * 16), 0, 0);
  for (int i = tid; i < (128 * 32 * 4) / 16; i += 256)        // 16KB of mask
    __builtin_amdgcn_global_load_async_to_lds_b128(
        (gp128)(v4i_*)(mc + (size_t)i * 16),
        (lp128)(v4i_*)((char*)Msh + (size_t)i * 16), 0, 0);
  __builtin_amdgcn_s_wait_asynccnt(0);
#else
  {
    const v8bf* ksrc = (const v8bf*)kc;
    v8bf* kdst = (v8bf*)Ksh;
    for (int i = tid; i < N_ * DK_ / 8; i += 256) kdst[i] = ksrc[i];
    const v8bf* vsrc = (const v8bf*)vc;
    v8bf* vdst = (v8bf*)Vsh;
    for (int i = tid; i < DK_ * N_ / 8; i += 256) vdst[i] = vsrc[i];
    const unsigned* msrc = (const unsigned*)mc;
    for (int i = tid; i < 128 * 32; i += 256) Msh[i] = msrc[i];
  }
#endif
  __syncthreads();

  int wv = tid >> 5, lane = tid & 31;
  int hlf = lane >> 4, n = lane & 15;
  int qrow0 = qbase + wv * 16;

  // Q A-operand for this wave's 16 rows (K dim = DK = 32: single operand)
  v16bf aq = load_a(Qg + ((size_t)bh * N_ + qrow0) * DK_, DK_, lane);

  v8f o0 = {0, 0, 0, 0, 0, 0, 0, 0};
  v8f o1 = {0, 0, 0, 0, 0, 0, 0, 0};
  float mrow[8], lrow[8];
#pragma unroll
  for (int g = 0; g < 8; ++g) { mrow[g] = -__builtin_inff(); lrow[g] = 0.0f; }
  bf16* Pw = Psh + wv * 16 * 32;
  const v8f zf = {0, 0, 0, 0, 0, 0, 0, 0};

  for (int kt = 0; kt < N_ / 32; ++kt) {     // 32 keys per step
    v16bf kb0 = load_b(Ksh + (size_t)(kt * 32) * DK_, DK_, lane);
    v16bf kb1 = load_b(Ksh + (size_t)(kt * 32 + 16) * DK_, DK_, lane);
    v8f s0 = wmma_bf(aq, kb0, zf);           // cols = keys kt*32 + n
    v8f s1 = wmma_bf(aq, kb1, zf);           // cols = keys kt*32 + 16 + n
#pragma unroll
    for (int g = 0; g < 8; ++g) {
      int r = wv * 16 + g + hlf * 8;
      unsigned mw = Msh[r * 32 + kt];
      float x0 = ((mw >> n) & 1u) ? s0[g] : -1e9f;
      float x1 = ((mw >> (16 + n)) & 1u) ? s1[g] : -1e9f;
      float tmax = bfly_max(fmaxf(x0, x1));
      float mnew = fmaxf(mrow[g], tmax);
      float corr = __expf(mrow[g] - mnew);
      float p0 = __expf(x0 - mnew);
      float p1 = __expf(x1 - mnew);
      float rs = bfly_add(p0 + p1);
      lrow[g] = lrow[g] * corr + rs;
      mrow[g] = mnew;
      o0[g] = o0[g] * corr;
      o1[g] = o1[g] * corr;
      // C-layout -> A-layout re-shuffle via wave-private LDS bounce
      Pw[(g + hlf * 8) * 32 + n]      = (bf16)p0;
      Pw[(g + hlf * 8) * 32 + 16 + n] = (bf16)p1;
    }
    asm volatile("s_wait_dscnt 0" ::: "memory");  // P stores visible in-wave
    v16bf ap  = load_a(Pw, 32, lane);
    v16bf vb0 = load_b(Vsh + (size_t)kt * 32, N_, lane);             // dk 0..15
    v16bf vb1 = load_b(Vsh + (size_t)16 * N_ + kt * 32, N_, lane);   // dk 16..31
    o0 = wmma_bf(ap, vb0, o0);
    o1 = wmma_bf(ap, vb1, o1);
  }

#pragma unroll
  for (int g = 0; g < 8; ++g) {
    float inv = 1.0f / lrow[g];
    int row = qrow0 + g + hlf * 8;
    size_t base = ((size_t)b * N_ + row) * HDK_ + hd * DK_;
    float y0 = o0[g] * inv, y1 = o1[g] * inv;
    Of[base + n] = y0;       Of[base + 16 + n] = y1;
    Ob[base + n] = (bf16)y0; Ob[base + 16 + n] = (bf16)y1;
  }
}

// ---------------------------------------------------------------------------
// GEMM + bias + residual + LayerNorm (Ncol fixed = 128).
// One wave computes a full 16x128 row block so LN reductions stay in-wave.
// ---------------------------------------------------------------------------
__global__ __launch_bounds__(256) void k_gemm_ln(
    const bf16* __restrict__ A, const bf16* __restrict__ Bt,
    const float* __restrict__ bias, const float* __restrict__ R,
    const float* __restrict__ gamma, const float* __restrict__ beta,
    float* __restrict__ Hf, bf16* __restrict__ Hb, int M, int Ka) {
  int wid = blockIdx.x * 8 + (threadIdx.x >> 5);
  int lane = threadIdx.x & 31;
  int m0 = wid * 16;
  if (m0 >= M) return;
  v8f c[8];
  const v8f zf = {0, 0, 0, 0, 0, 0, 0, 0};
#pragma unroll
  for (int t = 0; t < 8; ++t) c[t] = zf;
  for (int k0 = 0; k0 < Ka; k0 += 32) {
    v16bf a = load_a(A + (size_t)m0 * Ka + k0, Ka, lane);
#pragma unroll
    for (int t = 0; t < 8; ++t) {
      v16bf b = load_b(Bt + (size_t)t * 16 * Ka + k0, Ka, lane);
      c[t] = wmma_bf(a, b, c[t]);
    }
  }
  int hlf = lane >> 4, n = lane & 15;
#pragma unroll
  for (int t = 0; t < 8; ++t) {
    int col = t * 16 + n;
    float bv = bias ? bias[col] : 0.0f;
#pragma unroll
    for (int g = 0; g < 8; ++g) {
      int row = m0 + g + hlf * 8;
      c[t][g] = c[t][g] + bv + R[(size_t)row * D_ + col];
    }
  }
#pragma unroll
  for (int g = 0; g < 8; ++g) {
    float s = 0.0f;
#pragma unroll
    for (int t = 0; t < 8; ++t) s += c[t][g];
    float mean = bfly_add(s) * (1.0f / 128.0f);
    float vv = 0.0f;
#pragma unroll
    for (int t = 0; t < 8; ++t) { float d = c[t][g] - mean; vv += d * d; }
    float rstd = rsqrtf(bfly_add(vv) * (1.0f / 128.0f) + 1e-6f);
    int row = m0 + g + hlf * 8;
#pragma unroll
    for (int t = 0; t < 8; ++t) {
      int col = t * 16 + n;
      float o = (c[t][g] - mean) * rstd * gamma[col] + beta[col];
      Hf[(size_t)row * D_ + col] = o;
      if (Hb) Hb[(size_t)row * D_ + col] = (bf16)o;
    }
  }
}

// ---------------------------------------------------------------------------
// Host orchestration
// ---------------------------------------------------------------------------
extern "C" void kernel_launch(void* const* d_in, const int* in_sizes, int n_in,
                              void* d_out, int out_size, void* d_ws,
                              size_t ws_size, hipStream_t stream) {
  (void)in_sizes; (void)n_in; (void)out_size; (void)ws_size;
  const float* x   = (const float*)d_in[0];
  const int*   adj = (const int*)d_in[1];
  const float* W0  = (const float*)d_in[2];
  const float* b0  = (const float*)d_in[3];
  const float* Wq  = (const float*)d_in[4];
  const float* Wk  = (const float*)d_in[5];
  const float* Wv  = (const float*)d_in[6];
  const float* Wo  = (const float*)d_in[7];
  const float* g1  = (const float*)d_in[8];
  const float* be1 = (const float*)d_in[9];
  const float* W1  = (const float*)d_in[10];
  const float* b1  = (const float*)d_in[11];
  const float* W2  = (const float*)d_in[12];
  const float* b2  = (const float*)d_in[13];
  const float* g2  = (const float*)d_in[14];
  const float* be2 = (const float*)d_in[15];

  char* ws = (char*)d_ws;
  size_t off = 0;
  auto alloc = [&](size_t bytes) -> char* {
    char* p = ws + off;
    off += (bytes + 255) & ~(size_t)255;
    return p;
  };
  bf16*     xb  = (bf16*)alloc((size_t)M_ * DI_ * 2);
  bf16*     W0T = (bf16*)alloc((size_t)D_ * DI_ * 2);
  bf16*     WqT = (bf16*)alloc((size_t)L_ * HDK_ * D_ * 2);
  bf16*     WkT = (bf16*)alloc((size_t)L_ * HDK_ * D_ * 2);
  bf16*     WvT = (bf16*)alloc((size_t)L_ * HDK_ * D_ * 2);
  bf16*     WoT = (bf16*)alloc((size_t)L_ * D_ * HDK_ * 2);
  bf16*     W1T = (bf16*)alloc((size_t)L_ * D_ * D_ * 2);
  bf16*     W2T = (bf16*)alloc((size_t)L_ * D_ * D_ * 2);
  unsigned* Mg  = (unsigned*)alloc((size_t)B_ * N_ * (N_ / 32) * 4);
  float*    h   = (float*)alloc((size_t)M_ * D_ * 4);
  bf16*     hb  = (bf16*)alloc((size_t)M_ * D_ * 2);
  float*    h2  = (float*)alloc((size_t)M_ * D_ * 4);
  bf16*     h2b = (bf16*)alloc((size_t)M_ * D_ * 2);
  bf16*     Qb  = (bf16*)alloc((size_t)M_ * HDK_ * 2);
  bf16*     Kbv = (bf16*)alloc((size_t)M_ * HDK_ * 2);
  bf16*     Vt  = (bf16*)alloc((size_t)M_ * HDK_ * 2);
  float*    Of  = (float*)alloc((size_t)M_ * HDK_ * 4);
  bf16*     Ob  = (bf16*)alloc((size_t)M_ * HDK_ * 2);
  bf16*     tb  = (bf16*)alloc((size_t)M_ * D_ * 2);

  (void)hipFuncSetAttribute(reinterpret_cast<const void*>(k_attn),
                            hipFuncAttributeMaxDynamicSharedMemorySize,
                            ATT_LDS_BYTES);

  // ---- one-time prep ----
  k_cvt_bf16<<<(M_ * DI_ + 255) / 256, 256, 0, stream>>>(x, xb, M_ * DI_);
  k_trans_bf16<<<(DI_ * D_ + 255) / 256, 256, 0, stream>>>(W0, W0T, DI_, D_);
  for (int l = 0; l < L_; ++l) {
    size_t wqo = (size_t)l * D_ * HDK_;
    size_t wfo = (size_t)l * D_ * D_;
    k_trans_bf16<<<(D_ * HDK_ + 255) / 256, 256, 0, stream>>>(Wq + wqo, WqT + wqo, D_, HDK_);
    k_trans_bf16<<<(D_ * HDK_ + 255) / 256, 256, 0, stream>>>(Wk + wqo, WkT + wqo, D_, HDK_);
    k_trans_bf16<<<(D_ * HDK_ + 255) / 256, 256, 0, stream>>>(Wv + wqo, WvT + wqo, D_, HDK_);
    k_trans_bf16<<<(HDK_ * D_ + 255) / 256, 256, 0, stream>>>(Wo + wqo, WoT + wqo, HDK_, D_);
    k_trans_bf16<<<(D_ * D_ + 255) / 256, 256, 0, stream>>>(W1 + wfo, W1T + wfo, D_, D_);
    k_trans_bf16<<<(D_ * D_ + 255) / 256, 256, 0, stream>>>(W2 + wfo, W2T + wfo, D_, D_);
  }
  {
    int mwords = B_ * N_ * (N_ / 32);
    k_pack_mask<<<(mwords + 255) / 256, 256, 0, stream>>>(adj, Mg, mwords);
  }

  // ---- input projection: h = x @ W0 + b0 ----
  {
    int waves = (M_ / 16) * (D_ / 64);
    k_gemm_bias<<<waves / 8, 256, 0, stream>>>(xb, W0T, b0, h, hb,
                                               M_, DI_, D_, 0);
  }

  // ---- 6 GAT layers ----
  for (int l = 0; l < L_; ++l) {
    size_t wqo = (size_t)l * D_ * HDK_;
    size_t wfo = (size_t)l * D_ * D_;

    // fused q/k/v projection
    {
      int waves = (M_ / 16) * (HDK_ / 16);   // 8192
      k_qkv<<<waves / 8, 256, 0, stream>>>(
          hb, WqT + wqo, WkT + wqo, WvT + wqo, Qb, Kbv, Vt);
    }

    k_attn<<<B_ * H_ * (N_ / 128), 256, ATT_LDS_BYTES, stream>>>(
        Qb, Kbv, Vt, Mg, Of, Ob);

    // h2 = LN(o @ Wo + h)   (out-proj is bias-free)
    k_gemm_ln<<<(M_ / 16) / 8, 256, 0, stream>>>(
        Ob, WoT + wqo, nullptr, h, g1 + l * D_, be1 + l * D_,
        h2, h2b, M_, HDK_);

    // t = relu(h2 @ W1 + b1)
    {
      int waves = (M_ / 16) * (D_ / 64);
      k_gemm_bias<<<waves / 8, 256, 0, stream>>>(
          h2b, W1T + wfo, b1 + l * D_, nullptr, tb, M_, D_, D_, 1);
    }

    // h = LN(t @ W2 + b2 + h2)  (last layer writes d_out)
    float* hout = (l == L_ - 1) ? (float*)d_out : h;
    k_gemm_ln<<<(M_ / 16) / 8, 256, 0, stream>>>(
        tb, W2T + wfo, b2 + l * D_, h2, g2 + l * D_, be2 + l * D_,
        hout, hb, M_, D_);
  }
}
